// LinearModel_37297495998710
// MI455X (gfx1250) — 8-bit matrix operand order repaired and checked by reading, compiled, not run on hardware
//
#include <hip/hip_runtime.h>
#include <stdint.h>

typedef __attribute__((ext_vector_type(8))) int v8i;

static constexpr int B_ROWS = 65536;
static constexpr int D_IN   = 784;
static constexpr int K1_PAD = 832;
static constexpr int KT1    = 13;
static constexpr int H      = 128;
static constexpr int KT2    = 2;
static constexpr int D_OUT  = 10;

static constexpr size_t WS_QIN    = 0;
static constexpr size_t WS_Q1     = 54525952;
static constexpr size_t WS_Q2     = 62914560;
static constexpr size_t WS_B1     = 71303168;
static constexpr size_t WS_B2     = 71409664;
static constexpr size_t WS_B3     = 71426048;
static constexpr size_t WS_SCALES = 71428096;

__global__ void absmean3_kernel(const float* __restrict__ w1,
                                const float* __restrict__ w2,
                                const float* __restrict__ w3,
                                float* __restrict__ scales) {
    __shared__ float sm[256];
    const float* w; int n;
    if (blockIdx.x == 0)      { w = w1; n = D_IN * H; }
    else if (blockIdx.x == 1) { w = w2; n = H * H;    }
    else                      { w = w3; n = H * D_OUT;}
    float s = 0.0f;
    for (int i = threadIdx.x; i < n; i += 256) s += fabsf(w[i]);
    sm[threadIdx.x] = s;
    __syncthreads();
    for (int off = 128; off > 0; off >>= 1) {
        if ((int)threadIdx.x < off) sm[threadIdx.x] += sm[threadIdx.x + off];
        __syncthreads();
    }
    if (threadIdx.x == 0) scales[blockIdx.x] = sm[0] / (float)n;
}

__global__ void pack_sign_kernel(const float* __restrict__ w,
                                 int Kact, int Nact, int KT, int NT,
                                 uint32_t* __restrict__ out) {
    int idx = blockIdx.x * blockDim.x + threadIdx.x;
    int total = KT * NT * 256;
    if (idx >= total) return;
    int v    = idx & 7;
    int l    = (idx >> 3) & 31;
    int tile = idx >> 8;
    int nt   = tile % NT;
    int kt   = tile / NT;
    int kbase = (v >> 1) * 16 + ((l >= 16) ? 8 : 0) + (v & 1) * 4;
    int ng    = nt * 16 + (l & 15);
    uint32_t pack = 0;
    for (int b = 0; b < 4; ++b) {
        int kg = kt * 64 + kbase + b;
        int8_t sv = 0;
        if (kg < Kact && ng < Nact) {
            float x = w[(size_t)kg * Nact + ng];
            sv = (x > 0.0f) ? (int8_t)1 : ((x < 0.0f) ? (int8_t)-1 : (int8_t)0);
        }
        pack |= ((uint32_t)(uint8_t)sv) << (8 * b);
    }
    out[idx] = pack;
}

__global__ void quant_input_kernel(const float* __restrict__ x,
                                   const float* __restrict__ s_in_p,
                                   float* __restrict__ h0,
                                   int8_t* __restrict__ qin) {
    const float s   = *s_in_p;
    const float inv = 1.0f / s;
    long long tid = (long long)blockIdx.x * blockDim.x + threadIdx.x;
    int row = (int)(tid / (K1_PAD / 4));
    int c4  = (int)(tid % (K1_PAD / 4));
    if (row >= B_ROWS) return;
    int k = c4 * 4;
    if (k < D_IN) {
        const float4 xv = *(const float4*)(x + (size_t)row * D_IN + k);
        float q0 = fminf(fmaxf(rintf(xv.x * inv), -128.0f), 127.0f);
        float q1 = fminf(fmaxf(rintf(xv.y * inv), -128.0f), 127.0f);
        float q2 = fminf(fmaxf(rintf(xv.z * inv), -128.0f), 127.0f);
        float q3 = fminf(fmaxf(rintf(xv.w * inv), -128.0f), 127.0f);
        uchar4 qp;
        qp.x = (uint8_t)(int8_t)q0; qp.y = (uint8_t)(int8_t)q1;
        qp.z = (uint8_t)(int8_t)q2; qp.w = (uint8_t)(int8_t)q3;
        *(uchar4*)(qin + (size_t)row * K1_PAD + k) = qp;
        float4 hv; hv.x = q0 * s; hv.y = q1 * s; hv.z = q2 * s; hv.w = q3 * s;
        *(float4*)(h0 + (size_t)row * D_IN + k) = hv;
    } else {
        *(uint32_t*)(qin + (size_t)row * K1_PAD + k) = 0u;
    }
}

template <int NT, int KT>
__global__ void gemm_bin_kernel(const uint8_t* __restrict__ A, int lda,
                                const uint32_t* __restrict__ Bp,
                                const float* __restrict__ bias,
                                const float* __restrict__ wscale_p,
                                const float* __restrict__ ascale_p,
                                const float* __restrict__ si_p,
                                const float* __restrict__ so_p,
                                int act_mode,
                                float* __restrict__ outF, int ldo, int Nact,
                                float* __restrict__ outF2,
                                int8_t* __restrict__ qOut) {
    __shared__ uint32_t smemB[KT * NT * 256];

    const int tid = threadIdx.x;

    {
        constexpr int CHUNKS = KT * NT * 64;
        uint32_t lbase = (uint32_t)(uintptr_t)&smemB[0];
        uint64_t gbase = (uint64_t)(uintptr_t)Bp;
        for (int c = tid; c < CHUNKS; c += 256) {
            uint32_t laddr = lbase + (uint32_t)c * 16u;
            uint64_t gaddr = gbase + (uint64_t)c * 16u;
            asm volatile("global_load_async_to_lds_b128 %0, %1, off"
                         :: "v"(laddr), "v"(gaddr) : "memory");
        }
        asm volatile("s_wait_asynccnt 0x0" ::: "memory");
        __syncthreads();
    }

    const int lane  = tid & 31;
    const int wave  = blockIdx.x * (blockDim.x >> 5) + (tid >> 5);
    const int row0  = wave * 16;
    const int m     = lane & 15;
    const int khalf = lane >> 4;

    v8i acc[NT];
#pragma unroll
    for (int nt = 0; nt < NT; ++nt) acc[nt] = (v8i){0, 0, 0, 0, 0, 0, 0, 0};

    const uint8_t* arow = A + (size_t)(row0 + m) * lda + khalf * 8;

    auto loadB = [&](int idx) -> v8i {
        const uint4* p = (const uint4*)&smemB[((size_t)idx * 32 + lane) * 8];
        uint4 b0 = p[0];
        uint4 b1 = p[1];
        v8i bf;
        bf[0] = (int)b0.x; bf[1] = (int)b0.y; bf[2] = (int)b0.z; bf[3] = (int)b0.w;
        bf[4] = (int)b1.x; bf[5] = (int)b1.y; bf[6] = (int)b1.z; bf[7] = (int)b1.w;
        return bf;
    };

    uint2 a0 = *(const uint2*)(arow + 0);
    uint2 a1 = *(const uint2*)(arow + 16);
    uint2 a2 = *(const uint2*)(arow + 32);
    uint2 a3 = *(const uint2*)(arow + 48);

    for (int kt = 0; kt < KT; ++kt) {
        uint2 n0 = {0, 0}, n1 = {0, 0}, n2 = {0, 0}, n3 = {0, 0};
        if (kt + 1 < KT) {
            const uint8_t* ap = arow + (kt + 1) * 64;
            n0 = *(const uint2*)(ap + 0);
            n1 = *(const uint2*)(ap + 16);
            n2 = *(const uint2*)(ap + 32);
            n3 = *(const uint2*)(ap + 48);
        }
        v8i af;
        af[0] = (int)a0.x; af[1] = (int)a0.y;
        af[2] = (int)a1.x; af[3] = (int)a1.y;
        af[4] = (int)a2.x; af[5] = (int)a2.y;
        af[6] = (int)a3.x; af[7] = (int)a3.y;

        v8i bfs[NT];
#pragma unroll
        for (int nt = 0; nt < NT; ++nt) bfs[nt] = loadB(kt * NT + nt);

#pragma unroll
        for (int nt = 0; nt < NT; ++nt) {
            acc[nt] = __builtin_amdgcn_wmma_i32_16x16x64_iu8(
                true, af, true, bfs[nt], acc[nt], false, false);
        }
        a0 = n0; a1 = n1; a2 = n2; a3 = n3;
    }

    const float alpha  = (*ascale_p) * (*wscale_p);
    const float si     = *si_p;
    const float so     = *so_p;
    const float inv_si = 1.0f / si;
    const float inv_so = 1.0f / so;
    const int   n0c    = lane & 15;
    const int   mbase  = (lane >= 16) ? 8 : 0;

#pragma unroll
    for (int nt = 0; nt < NT; ++nt) {
        int ncol = nt * 16 + n0c;
        if (ncol < Nact) {
            float bn = bias[ncol];
#pragma unroll
            for (int g = 0; g < 8; ++g) {
                int row = row0 + mbase + g;
                float v = alpha * (float)acc[nt][g] + bn;
                float qa = fminf(fmaxf(rintf(v * inv_si), -128.0f), 127.0f);
                v = qa * si;
                v = (act_mode == 0) ? fmaxf(v, 0.0f) : (1.0f / (1.0f + expf(-v)));
                float qb = fminf(fmaxf(rintf(v * inv_so), -128.0f), 127.0f);
                v = qb * so;
                outF[(size_t)row * ldo + ncol] = v;
                if (outF2) outF2[(size_t)row * ldo + ncol] = v;
                if (qOut)  qOut[(size_t)row * ldo + ncol] = (int8_t)qb;
            }
        }
    }
}

extern "C" void kernel_launch(void* const* d_in, const int* in_sizes, int n_in,
                              void* d_out, int out_size, void* d_ws, size_t ws_size,
                              hipStream_t stream) {
    const float* input = (const float*)d_in[0];
    const float* w1 = (const float*)d_in[1];
    const float* b1 = (const float*)d_in[2];
    const float* w2 = (const float*)d_in[3];
    const float* b2 = (const float*)d_in[4];
    const float* w3 = (const float*)d_in[5];
    const float* b3 = (const float*)d_in[6];
    const float* s_in = (const float*)d_in[7];
    const float* s1i  = (const float*)d_in[8];
    const float* s1o  = (const float*)d_in[9];
    const float* s2i  = (const float*)d_in[10];
    const float* s2o  = (const float*)d_in[11];
    const float* s3i  = (const float*)d_in[12];
    const float* s3o  = (const float*)d_in[13];

    float* out    = (float*)d_out;
    float* out_x  = out;
    float* out_h0 = out_x  + (size_t)B_ROWS * D_OUT;
    float* out_h1 = out_h0 + (size_t)B_ROWS * D_IN;
    float* out_h2 = out_h1 + (size_t)B_ROWS * H;
    float* out_h3 = out_h2 + (size_t)B_ROWS * H;

    uint8_t*  ws     = (uint8_t*)d_ws;
    int8_t*   qin    = (int8_t*)(ws + WS_QIN);
    int8_t*   q1     = (int8_t*)(ws + WS_Q1);
    int8_t*   q2     = (int8_t*)(ws + WS_Q2);
    uint32_t* B1p    = (uint32_t*)(ws + WS_B1);
    uint32_t* B2p    = (uint32_t*)(ws + WS_B2);
    uint32_t* B3p    = (uint32_t*)(ws + WS_B3);
    float*    scales = (float*)(ws + WS_SCALES);

    absmean3_kernel<<<3, 256, 0, stream>>>(w1, w2, w3, scales);
    pack_sign_kernel<<<(KT1 * 8 * 256 + 255) / 256, 256, 0, stream>>>(w1, D_IN, H, KT1, 8, B1p);
    pack_sign_kernel<<<(KT2 * 8 * 256 + 255) / 256, 256, 0, stream>>>(w2, H, H, KT2, 8, B2p);
    pack_sign_kernel<<<(KT2 * 1 * 256 + 255) / 256, 256, 0, stream>>>(w3, H, D_OUT, KT2, 1, B3p);

    {
        long long threads = (long long)B_ROWS * (K1_PAD / 4);
        int blocks = (int)((threads + 255) / 256);
        quant_input_kernel<<<blocks, 256, 0, stream>>>(input, s_in, out_h0, qin);
    }

    const int gemm_blocks = (B_ROWS / 16) / 8;

    gemm_bin_kernel<8, KT1><<<gemm_blocks, 256, 0, stream>>>(
        (const uint8_t*)qin, K1_PAD, B1p,
        b1, scales + 0, s_in, s1i, s1o,  0,
        out_h1, H, H, nullptr, q1);

    gemm_bin_kernel<8, KT2><<<gemm_blocks, 256, 0, stream>>>(
        (const uint8_t*)q1, H, B2p,
        b2, scales + 1, s1o, s2i, s2o,  0,
        out_h2, H, H, nullptr, q2);

    gemm_bin_kernel<1, KT2><<<gemm_blocks, 256, 0, stream>>>(
        (const uint8_t*)q2, H, B3p,
        b3, scales + 2, s2o, s3i, s3o,  1,
        out_x, D_OUT, D_OUT, out_h3, nullptr);
}
